// DirectedHebbianGraph_14508399526341
// MI455X (gfx1250) — matrix operand order repaired and checked by reading, not compiled, not run on hardware
//
#include <hip/hip_runtime.h>
#include <math.h>

// ---------------------------------------------------------------------------
// Problem constants (from reference): B=4096, D=H0=H1=O=1024
// ---------------------------------------------------------------------------
#define BB   4096
#define DD   1024
#define OO   1024
#define ALPHA_C 0.3f
#define CLAMP_C 0.3f

typedef __attribute__((ext_vector_type(16))) __bf16 v16bf;
typedef __attribute__((ext_vector_type(8)))  __bf16 v8bf;
typedef __attribute__((ext_vector_type(8)))  float  v8f;
typedef __attribute__((ext_vector_type(4)))  int    v4i;

#define AS_GLOBAL __attribute__((address_space(1)))
#define AS_LDS    __attribute__((address_space(3)))

#ifndef __has_builtin
#define __has_builtin(x) 0
#endif
#if __has_builtin(__builtin_amdgcn_global_load_async_to_lds_b128) && \
    __has_builtin(__builtin_amdgcn_s_wait_asynccnt)
#define ASYNC_OK 1
#else
#define ASYNC_OK 0
#endif

// ---------------------------------------------------------------------------
// Elementwise helpers
// ---------------------------------------------------------------------------
__global__ __launch_bounds__(256) void cast_f32_to_bf16(const float* __restrict__ in,
                                                        __bf16* __restrict__ out, int n) {
  int i = blockIdx.x * 256 + threadIdx.x;
  if (i < n) out[i] = (__bf16)in[i];
}

// weff = bf16(w + W0 * heb)
__global__ __launch_bounds__(256) void make_weff(const float* __restrict__ w,
                                                 const float* __restrict__ heb,
                                                 const float* __restrict__ Wp,
                                                 __bf16* __restrict__ out, int n) {
  int i = blockIdx.x * 256 + threadIdx.x;
  if (i < n) out[i] = (__bf16)(w[i] + Wp[0] * heb[i]);
}

__global__ __launch_bounds__(256) void relu_cast_bf16(const float* __restrict__ in,
                                                      __bf16* __restrict__ out, int n) {
  int i = blockIdx.x * 256 + threadIdx.x;
  if (i < n) out[i] = (__bf16)fmaxf(in[i], 0.0f);
}

// y (4096x1024 f32) -> y_out (cols 0..1021) + tanh scratch for cols 1022 (W) / 1021 (dopa)
__global__ __launch_bounds__(256) void finalize_y(const float* __restrict__ y,
                                                  float* __restrict__ yout,
                                                  float* __restrict__ t0,
                                                  float* __restrict__ t1) {
  int i = blockIdx.x * 256 + threadIdx.x;
  if (i >= BB * OO) return;
  int b = i / OO, c = i % OO;
  float v = y[i];
  if (c < OO - 2) yout[(size_t)b * (OO - 2) + c] = v;
  if (c == OO - 2) t0[b] = tanhf(v);
  if (c == OO - 3) t1[b] = tanhf(v);
}

// Deterministic single-block tree reduction of the two tanh columns -> means
__global__ __launch_bounds__(256) void reduce_means(const float* __restrict__ t0,
                                                    const float* __restrict__ t1,
                                                    float* __restrict__ out,
                                                    size_t posW, size_t posD) {
  __shared__ float s0[256], s1[256];
  int tid = threadIdx.x;
  float a = 0.f, b = 0.f;
  for (int i = tid; i < BB; i += 256) { a += t0[i]; b += t1[i]; }
  s0[tid] = a; s1[tid] = b;
  __syncthreads();
  for (int s = 128; s > 0; s >>= 1) {
    if (tid < s) { s0[tid] += s0[tid + s]; s1[tid] += s1[tid + s]; }
    __syncthreads();
  }
  if (tid == 0) { out[posW] = s0[0] / (float)BB; out[posD] = s1[0] / (float)BB; }
}

// ---------------------------------------------------------------------------
// Tiled bf16 WMMA GEMM:  C(MxN) = op(A) @ B,  f32 accumulate
//   TRANS_A = false: A is (M x K) row-major, lda = K-stride
//   TRANS_A = true : A is (K x M) row-major (computes A^T @ B), lda = M-stride
//   B is (K x N) row-major.
// Epilogues:
//   EPI 0: C = acc (f32), Cbf = bf16(acc)
//   EPI 1: C += acc (f32), Cbf = bf16(acc)
//   EPI 2: C = clamp(aux + ALPHA*acc, +-CLAMP)   (Hebbian update, straight to d_out)
// Tiling: 128x128 block tile, BK=32; 8 waves (wave32) in a 2x4 grid,
//         each wave computes 64x32 via 4x2 v_wmma_f32_16x16x32_bf16 tiles.
// Pipeline per K-step:
//   1) issue next tile's global loads into VGPRs + async A-tile copies (ASYNCcnt)
//   2) 8 WMMAs on current LDS buffer
//   3) scatter the register-staged tiles into LDS (loadcnt wait lands *after* WMMAs)
//   4) s_wait_asynccnt + barrier, flip buffers
// ---------------------------------------------------------------------------
#define BM 128
#define BN 128
#define BK 32
#define LDSS 40   // padded halfword stride (40*2=80B, 16B-aligned, avoids bank conflicts)

template <bool TRANS_A, int EPI>
__global__ __launch_bounds__(256) void gemm_bf16_wmma(
    const __bf16* __restrict__ A, int lda,
    const __bf16* __restrict__ B, int ldb,
    float* __restrict__ C, int ldc,
    __bf16* __restrict__ Cbf,
    const float* __restrict__ aux,
    int M, int N, int K) {
  __shared__ __bf16 As[2][BM * LDSS];
  __shared__ __bf16 Bs[2][BN * LDSS];

  const int tid  = threadIdx.x;
  const int wave = tid >> 5;
  const int lane = tid & 31;
  const int l16  = lane & 15;
  const int lhi  = lane >> 4;          // 0 or 1
  const int wm   = (wave >> 2) * 64;   // wave M offset in tile: 0 / 64
  const int wn   = (wave & 3) * 32;    // wave N offset in tile: 0/32/64/96
  const int m0   = blockIdx.y * BM;
  const int n0   = blockIdx.x * BN;

  v8f acc[4][2];
#pragma unroll
  for (int t = 0; t < 4; ++t)
#pragma unroll
    for (int u = 0; u < 2; ++u)
#pragma unroll
      for (int v = 0; v < 8; ++v) acc[t][u][v] = 0.0f;

  v8bf aReg[2];  // register staging for transpose-scatter tiles
  v8bf bReg[2];

  // ---- phase 1a: issue global loads for tiles that need an LDS transpose ----
  auto loadTiles = [&](int k0) {
    if constexpr (TRANS_A) {
#pragma unroll
      for (int c = 0; c < 2; ++c) {
        int linear = tid * 16 + c * 8;
        int m  = linear & 127;              // contiguous along global row (coalesced)
        int kk = linear >> 7;               // 0..31
        const __bf16* src = A + (size_t)(k0 + kk) * lda + m0 + m;
        aReg[c] = *(const v8bf*)src;
        __builtin_prefetch(src + 2 * (size_t)BK * lda, 0, 3);
      }
    }
#pragma unroll
    for (int c = 0; c < 2; ++c) {
      int linear = tid * 16 + c * 8;
      int n  = linear & 127;
      int kk = linear >> 7;
      const __bf16* src = B + (size_t)(k0 + kk) * ldb + n0 + n;
      bReg[c] = *(const v8bf*)src;
      __builtin_prefetch(src + 2 * (size_t)BK * ldb, 0, 3);
    }
  };

  // ---- phase 1b: async global->LDS copy of the A tile (layout already matches) ----
  auto stageAsyncA = [&](int buf, int k0) {
    if constexpr (!TRANS_A) {
#pragma unroll
      for (int c = 0; c < 2; ++c) {
        int linear = tid * 16 + c * 8;      // 128x32 elems / 256 thr = 16 per thr
        int row = linear >> 5;
        int col = linear & 31;
        const __bf16* src = A + (size_t)(m0 + row) * lda + k0 + col;
        __bf16* dst = &As[buf][row * LDSS + col];
#if ASYNC_OK
        __builtin_amdgcn_global_load_async_to_lds_b128(
            (AS_GLOBAL v4i*)(void*)src, (AS_LDS v4i*)(void*)dst, 0, 0);
#else
        *(v8bf*)dst = *(const v8bf*)src;
#endif
        __builtin_prefetch(src + 2 * BK, 0, 3);
      }
    }
  };

  // ---- phase 3: scatter register-staged tiles into LDS (transposed) ----
  auto scatterTiles = [&](int buf) {
    if constexpr (TRANS_A) {
#pragma unroll
      for (int c = 0; c < 2; ++c) {
        int linear = tid * 16 + c * 8;
        int m  = linear & 127;
        int kk = linear >> 7;
#pragma unroll
        for (int i = 0; i < 8; ++i) As[buf][(m + i) * LDSS + kk] = aReg[c][i];
      }
    }
#pragma unroll
    for (int c = 0; c < 2; ++c) {
      int linear = tid * 16 + c * 8;
      int n  = linear & 127;
      int kk = linear >> 7;
#pragma unroll
      for (int i = 0; i < 8; ++i) Bs[buf][(n + i) * LDSS + kk] = bReg[c][i];
    }
  };

  // ---- phase 2: 8 WMMAs on LDS buffer `buf` (ISA 7.12.2 fragment layouts) ----
  auto compute = [&](int buf) {
    v16bf afrag[4];
    const int akb = lhi * 8;
#pragma unroll
    for (int t = 0; t < 4; ++t) {
      int m = wm + t * 16 + l16;
      v8bf lo = *(const v8bf*)(&As[buf][m * LDSS + akb]);
      v8bf hi = *(const v8bf*)(&As[buf][m * LDSS + akb + 16]);
      afrag[t] = __builtin_shufflevector(lo, hi, 0, 1, 2, 3, 4, 5, 6, 7,
                                         8, 9, 10, 11, 12, 13, 14, 15);
    }
    v16bf bfrag[2];
    const int bkb = lhi * 8;
#pragma unroll
    for (int u = 0; u < 2; ++u) {
      int n = wn + u * 16 + l16;
      v8bf lo = *(const v8bf*)(&Bs[buf][n * LDSS + bkb]);
      v8bf hi = *(const v8bf*)(&Bs[buf][n * LDSS + bkb + 16]);
      bfrag[u] = __builtin_shufflevector(lo, hi, 0, 1, 2, 3, 4, 5, 6, 7,
                                         8, 9, 10, 11, 12, 13, 14, 15);
    }
#pragma unroll
    for (int t = 0; t < 4; ++t)
#pragma unroll
      for (int u = 0; u < 2; ++u)
        acc[t][u] = __builtin_amdgcn_wmma_f32_16x16x32_bf16(
            false, afrag[t], false, bfrag[u], (short)0, acc[t][u], false, false);
  };

  // ---- software pipeline ----
  const int nIter = K / BK;
  loadTiles(0);
  stageAsyncA(0, 0);
  scatterTiles(0);
#if ASYNC_OK
  if constexpr (!TRANS_A) __builtin_amdgcn_s_wait_asynccnt(0);
#endif
  __syncthreads();
  for (int it = 0; it < nIter; ++it) {
    const int buf = it & 1;
    const bool more = (it + 1 < nIter);
    if (more) {
      loadTiles((it + 1) * BK);            // global loads in flight during WMMAs
      stageAsyncA(buf ^ 1, (it + 1) * BK); // ASYNCcnt copies in flight during WMMAs
    }
    compute(buf);
    if (more) scatterTiles(buf ^ 1);       // loadcnt wait lands here, after WMMAs
#if ASYNC_OK
    if constexpr (!TRANS_A) __builtin_amdgcn_s_wait_asynccnt(0);
#endif
    __syncthreads();
  }

  // ---- epilogue: C/D layout (7.12.2): VGPR v -> M = v + 8*lhi; N = l16 ----
#pragma unroll
  for (int t = 0; t < 4; ++t)
#pragma unroll
    for (int u = 0; u < 2; ++u)
#pragma unroll
      for (int v = 0; v < 8; ++v) {
        int m = m0 + wm + t * 16 + lhi * 8 + v;
        int n = n0 + wn + u * 16 + l16;
        size_t idx = (size_t)m * ldc + n;
        float a = acc[t][u][v];
        if constexpr (EPI == 0) {
          C[idx] = a;
          Cbf[idx] = (__bf16)a;
        } else if constexpr (EPI == 1) {
          C[idx] += a;
          Cbf[idx] = (__bf16)a;
        } else {
          float r = aux[idx] + ALPHA_C * a;
          r = fminf(CLAMP_C, fmaxf(-CLAMP_C, r));
          C[idx] = r;
        }
      }
}

// ---------------------------------------------------------------------------
// Host-side orchestration
// ---------------------------------------------------------------------------
extern "C" void kernel_launch(void* const* d_in, const int* in_sizes, int n_in,
                              void* d_out, int out_size, void* d_ws, size_t ws_size,
                              hipStream_t stream) {
  (void)in_sizes; (void)n_in; (void)out_size; (void)ws_size;

  // inputs (setup_inputs order)
  const float* x        = (const float*)d_in[0];
  const float* w_m[6]   = {(const float*)d_in[1], (const float*)d_in[2],
                           (const float*)d_in[3], (const float*)d_in[4],
                           (const float*)d_in[5], (const float*)d_in[6]};
  const float* heb_m[6] = {(const float*)d_in[7], (const float*)d_in[8],
                           (const float*)d_in[9], (const float*)d_in[10],
                           (const float*)d_in[11], (const float*)d_in[12]};
  const float* Wscalar  = (const float*)d_in[13];

  // output layout: y_out(4096x1022) | 6x heb(1024x1024) | W_new | dopa
  float* out = (float*)d_out;
  const size_t Y_ELEMS = (size_t)BB * (OO - 2);
  const size_t HEB     = (size_t)DD * OO;
  float* yout   = out;
  float* hebo[6];
  for (int i = 0; i < 6; ++i) hebo[i] = out + Y_ELEMS + (size_t)i * HEB;
  const size_t posW = Y_ELEMS + 6 * HEB;
  const size_t posD = posW + 1;

  // workspace carve-up
  char* ws = (char*)d_ws;
  size_t off = 0;
  auto alloc = [&](size_t bytes) { char* p = ws + off; off += (bytes + 255) & ~(size_t)255; return p; };
  const size_t BDn = (size_t)BB * DD;       // 4096*1024
  __bf16* xbf   = (__bf16*)alloc(BDn * 2);
  __bf16* h0bf  = (__bf16*)alloc(BDn * 2);
  __bf16* h1bf  = (__bf16*)alloc(BDn * 2);
  __bf16* weff[6];
  for (int i = 0; i < 6; ++i) weff[i] = (__bf16*)alloc(HEB * 2);
  float*  h0x   = (float*)alloc(BDn * 4);   // h0 pre-activation
  float*  hx    = (float*)alloc(BDn * 4);   // h1_x then += h1_0
  float*  yacc  = (float*)alloc(BDn * 4);   // y_x + y_0 + y_1
  __bf16* h0x_bf = (__bf16*)alloc(BDn * 2); // products in bf16 for Hebbian B-operands
  __bf16* h1x_bf = (__bf16*)alloc(BDn * 2);
  __bf16* h10_bf = (__bf16*)alloc(BDn * 2);
  __bf16* yx_bf  = (__bf16*)alloc(BDn * 2);
  __bf16* y0_bf  = (__bf16*)alloc(BDn * 2);
  __bf16* y1_bf  = (__bf16*)alloc(BDn * 2);
  float*  t0    = (float*)alloc(BB * 4);
  float*  t1    = (float*)alloc(BB * 4);

  // 1) casts + effective weights
  cast_f32_to_bf16<<<(BDn + 255) / 256, 256, 0, stream>>>(x, xbf, (int)BDn);
  for (int i = 0; i < 6; ++i)
    make_weff<<<(HEB + 255) / 256, 256, 0, stream>>>(w_m[i], heb_m[i], Wscalar, weff[i], (int)HEB);

  const dim3 blk(256);
  const dim3 gFwd(OO / BN, BB / BM);   // (8, 32)
  const dim3 gHeb(OO / BN, DD / BM);   // (8, 8)

  // 2) forward GEMMs (M=4096, N=1024, K=1024)
  gemm_bf16_wmma<false, 0><<<gFwd, blk, 0, stream>>>(xbf, DD, weff[0], OO, h0x, OO, h0x_bf, nullptr, BB, OO, DD);
  relu_cast_bf16<<<(BDn + 255) / 256, 256, 0, stream>>>(h0x, h0bf, (int)BDn);

  gemm_bf16_wmma<false, 0><<<gFwd, blk, 0, stream>>>(xbf, DD, weff[1], OO, hx, OO, h1x_bf, nullptr, BB, OO, DD);
  gemm_bf16_wmma<false, 1><<<gFwd, blk, 0, stream>>>(h0bf, DD, weff[2], OO, hx, OO, h10_bf, nullptr, BB, OO, DD);
  relu_cast_bf16<<<(BDn + 255) / 256, 256, 0, stream>>>(hx, h1bf, (int)BDn);

  gemm_bf16_wmma<false, 0><<<gFwd, blk, 0, stream>>>(xbf, DD, weff[3], OO, yacc, OO, yx_bf, nullptr, BB, OO, DD);
  gemm_bf16_wmma<false, 1><<<gFwd, blk, 0, stream>>>(h0bf, DD, weff[4], OO, yacc, OO, y0_bf, nullptr, BB, OO, DD);
  gemm_bf16_wmma<false, 1><<<gFwd, blk, 0, stream>>>(h1bf, DD, weff[5], OO, yacc, OO, y1_bf, nullptr, BB, OO, DD);

  // 3) y epilogue + deterministic tanh means
  finalize_y<<<(BDn + 255) / 256, 256, 0, stream>>>(yacc, yout, t0, t1);
  reduce_means<<<1, 256, 0, stream>>>(t0, t1, out, posW, posD);

  // 4) Hebbian updates: heb_new = clamp(heb + ALPHA * act^T @ prod)  (M=N=1024, K=4096)
  gemm_bf16_wmma<true, 2><<<gHeb, blk, 0, stream>>>(xbf,  DD, h0x_bf, OO, hebo[0], OO, nullptr, heb_m[0], DD, OO, BB);
  gemm_bf16_wmma<true, 2><<<gHeb, blk, 0, stream>>>(xbf,  DD, h1x_bf, OO, hebo[1], OO, nullptr, heb_m[1], DD, OO, BB);
  gemm_bf16_wmma<true, 2><<<gHeb, blk, 0, stream>>>(h0bf, DD, h10_bf, OO, hebo[2], OO, nullptr, heb_m[2], DD, OO, BB);
  gemm_bf16_wmma<true, 2><<<gHeb, blk, 0, stream>>>(xbf,  DD, yx_bf,  OO, hebo[3], OO, nullptr, heb_m[3], DD, OO, BB);
  gemm_bf16_wmma<true, 2><<<gHeb, blk, 0, stream>>>(h0bf, DD, y0_bf,  OO, hebo[4], OO, nullptr, heb_m[4], DD, OO, BB);
  gemm_bf16_wmma<true, 2><<<gHeb, blk, 0, stream>>>(h1bf, DD, y1_bf,  OO, hebo[5], OO, nullptr, heb_m[5], DD, OO, BB);
}
